// SymmetricContraction_84859963834516
// MI455X (gfx1250) — compile-verified, matrix operand order checked by reading, not run on hardware
//
#include <hip/hip_runtime.h>

// ---------------------------------------------------------------------------
// Symmetric contraction (MACE-style), reformulated as a bf16 WMMA GEMM:
//   M (65536 x 4384 moments, built on the fly)  @  Ubig (4384 x 640)
// with the species/channel weight contraction fused into the epilogue.
// CDNA5 paths: v_wmma_f32_16x16x32_bf16 inner loop, double-buffered
// global_load_async_to_lds_b128 staging + s_wait_asynccnt, ds_load_b128
// fragment feeds, ds_add_f32 epilogue reduction.
// ---------------------------------------------------------------------------

typedef __attribute__((ext_vector_type(16))) __bf16 v16bf;
typedef __attribute__((ext_vector_type(8)))  float  v8f;

#define NROWS   65536          // N*C = 512*128
#define KP      4384           // 4096 (v^3) + 256 (v^2) + 16 (v) + 16 pad
#define KT      137            // KP / 32
#define NCOL    640            // 621 real columns + pad to 40 tiles of 16
#define NREAL   621
#define SLICE_BYTES (NCOL * 32 * 2)            // 40960 per K-step
#define UBIG_BYTES (KT * NCOL * 32 * 2)        // 5,611,520 (bf16, K-tiled)
#define WBIG_BYTES (64 * 128 * NCOL * 4)       // 20,971,520 (f32)

// column segments: 4x order-3, 4x order-2, 4x order-1
__constant__ int c_colbase[12] = {0,16,100,270, 522,525,540,570, 605,606,609,614};
__constant__ int c_mul[12]     = {16,28,34,36,  3,5,6,5,         1,1,1,1};
__constant__ int c_ir[12]      = {1,3,5,7,      1,3,5,7,         1,3,5,7};
__constant__ int c_ioff[12]    = {0,1,4,9,      0,1,4,9,         0,1,4,9};

struct Ptrs12 { const float* p[12]; };

__device__ __forceinline__ int seg_of(int col) {
  int s = 0;
#pragma unroll
  for (int t = 1; t < 12; ++t) s += (col >= c_colbase[t]) ? 1 : 0;
  return s;
}

__device__ __forceinline__ unsigned short f2bf(float f) {
  unsigned u = __builtin_bit_cast(unsigned, f);
  u += 0x7fffu + ((u >> 16) & 1u);          // round-to-nearest-even
  return (unsigned short)(u >> 16);
}

// Async global -> LDS DMA (CDNA5).  GVS addressing: mem = SADDR + VADDR_u32.
// VDST VGPR holds the LDS byte address; tracked with ASYNCcnt.
__device__ __forceinline__ void async_b128(unsigned lds_addr, unsigned goff,
                                           unsigned long long sbase) {
  asm volatile("global_load_async_to_lds_b128 %0, %1, %2"
               :: "v"(lds_addr), "v"(goff), "s"(sbase)
               : "memory");
}

// ---------------------------------------------------------------------------
// Prep: Ubig[kt][col][32] bf16 (K-tiled so the GEMM DMAs a contiguous 40KB
// slice per K-step).
// ---------------------------------------------------------------------------
__global__ void k_ubig(Ptrs12 P, unsigned short* __restrict__ ubig) {
  int idx = blockIdx.x * 256 + threadIdx.x;       // total = KT*NCOL*32
  int kl  = idx & 31;
  int rem = idx >> 5;
  int col = rem % NCOL;
  int kt  = rem / NCOL;
  int k   = kt * 32 + kl;
  float v = 0.0f;
  if (col < NREAL && k < 4368) {
    int s     = seg_of(col);
    int local = col - c_colbase[s];
    int ir = c_ir[s], mul = c_mul[s];
    int kk = local / ir, i = local % ir;
    if (s < 4)      { if (k < 4096)              v = P.p[s][(k * mul + kk) * ir + i]; }
    else if (s < 8) { if (k >= 4096 && k < 4352) v = P.p[s][((k - 4096) * mul + kk) * ir + i]; }
    else            { if (k >= 4352)             v = P.p[s][((k - 4352) * mul + kk) * ir + i]; }
  }
  ubig[idx] = f2bf(v);
}

// Wbig[(species*128 + c)*NCOL + col] = W_seg[species, k(col), c]
__global__ void k_wbig(Ptrs12 P, float* __restrict__ wbig) {
  int idx = blockIdx.x * 256 + threadIdx.x;       // total = 64*128*NCOL
  int col = idx % NCOL;
  int sc  = idx / NCOL;
  int sp  = sc / 128, c = sc % 128;
  float v = 0.0f;
  if (col < NREAL) {
    int s  = seg_of(col);
    int kk = (col - c_colbase[s]) / c_ir[s];
    v = P.p[s][(sp * c_mul[s] + kk) * 128 + c];
  }
  wbig[idx] = v;
}

// otab[col] = output irrep slot (0..15), 16 = dump slot for pad columns
__global__ void k_otab(int* __restrict__ otab) {
  int col = blockIdx.x * 256 + threadIdx.x;
  if (col >= NCOL) return;
  int slot = 16;
  if (col < NREAL) {
    int s = seg_of(col);
    slot  = c_ioff[s] + (col - c_colbase[s]) % c_ir[s];
  }
  otab[col] = slot;
}

// ---------------------------------------------------------------------------
// Main fused GEMM: 64 rows x 640 cols per workgroup.
// 512 threads = 16 waves: 4 row-strips x 4 col-strips, each wave owns a
// 16x160 tile = 10 accumulator fragments (80 VGPRs).
// ---------------------------------------------------------------------------
union FragU { uint4 q[2]; v16bf v; };

__global__ __launch_bounds__(512) void k_gemm(
    const float* __restrict__ x, const int* __restrict__ species,
    const unsigned short* __restrict__ ubig, const float* __restrict__ wbig,
    const int* __restrict__ otab, float* __restrict__ out)
{
  __shared__ __align__(16) unsigned short Bs[2][NCOL * 32]; // 2x40KB, col*32+kl
  __shared__ __align__(16) unsigned short Ms[64 * 32];      //  4KB, row*32+kl
  __shared__ float xs[64][16];
  __shared__ int   scs[64];
  __shared__ float outs[64 * 17];

  const int t      = threadIdx.x;
  const int lane   = t & 31;
  const int wave   = t >> 5;
  const int rstrip = wave >> 2;          // 0..3
  const int cstrip = wave & 3;           // 0..3
  const int half   = lane >> 4;          // 0 or 1 (wave32 half)
  const int l16    = lane & 15;
  const int r0     = blockIdx.x * 64;

  // kick off async DMA of the first B K-slice while we set up
  {
    unsigned long long gb = (unsigned long long)(const char*)ubig;
    unsigned ldsb = (unsigned)(size_t)&Bs[0][0];
#pragma unroll
    for (int q = 0; q < 5; ++q) {
      unsigned chunk = (unsigned)t + 512u * q;
      async_b128(ldsb + chunk * 16u, chunk * 16u, gb);
    }
  }

#pragma unroll
  for (int q = 0; q < 2; ++q) {          // 64 rows x 16 x-values
    int e = t + 512 * q;
    xs[e >> 4][e & 15] = x[(r0 + (e >> 4)) * 16 + (e & 15)];
  }
  if (t < 64) { int r = r0 + t; scs[t] = species[r >> 7] * 128 + (r & 127); }

  v8f acc[10];
#pragma unroll
  for (int i = 0; i < 10; ++i)
#pragma unroll
    for (int j = 0; j < 8; ++j) acc[i][j] = 0.0f;

  __syncthreads();   // xs/scs visible before first Ms build

  for (int kt = 0; kt < KT; ++kt) {
    const int cur = kt & 1;

    // ---- issue async DMA for the NEXT K-slice into the other buffer ----
    if (kt + 1 < KT) {
      unsigned long long gb =
          (unsigned long long)((const char*)ubig + (size_t)(kt + 1) * SLICE_BYTES);
      unsigned ldsb = (unsigned)(size_t)&Bs[cur ^ 1][0];
#pragma unroll
      for (int q = 0; q < 5; ++q) {
        unsigned chunk = (unsigned)t + 512u * q;
        async_b128(ldsb + chunk * 16u, chunk * 16u, gb);
      }
    }

    // ---- build 64x32 bf16 moment tile (overlaps with in-flight DMA) ----
#pragma unroll
    for (int q = 0; q < 4; ++q) {
      int e = t + 512 * q;
      int row = e >> 5, kl = e & 31;
      int kg = kt * 32 + kl;
      float v;
      if (kg < 4096)      v = xs[row][kg >> 8] * xs[row][(kg >> 4) & 15] * xs[row][kg & 15];
      else if (kg < 4352) { int kr = kg - 4096; v = xs[row][kr >> 4] * xs[row][kr & 15]; }
      else if (kg < 4368) v = xs[row][kg - 4352];
      else                v = 0.0f;
      Ms[row * 32 + kl] = f2bf(v);
    }

    // wait for *current* slice DMA (in-order completion; <=5 means the older
    // stage retired), then barrier so every wave sees LDS fully populated.
    if (kt + 1 < KT) asm volatile("s_wait_asynccnt 5" ::: "memory");
    else             asm volatile("s_wait_asynccnt 0" ::: "memory");
    __syncthreads();

    // ---- A fragment: lanes 0-15 hold K{0..7,16..23}, lanes 16-31 K{8..15,24..31}
    FragU fa;
    {
      const uint4* mrow = (const uint4*)&Ms[(rstrip * 16 + l16) * 32];
      fa.q[0] = mrow[half];
      fa.q[1] = mrow[half + 2];
    }
    // ---- 10 column tiles ----
#pragma unroll
    for (int tt = 0; tt < 10; ++tt) {
      int col = cstrip * 160 + tt * 16 + l16;
      FragU fb;
      const uint4* brow = (const uint4*)&Bs[cur][col * 32];
      fb.q[0] = brow[half];
      fb.q[1] = brow[half + 2];
      acc[tt] = __builtin_amdgcn_wmma_f32_16x16x32_bf16(
          false, fa.v, false, fb.v, (short)0, acc[tt], false, false);
    }
    __syncthreads();   // protect Ms (and retired B buffer) before overwrite
  }

  // ---- epilogue: out[r,i] = sum_col B[r,col] * Wbig[sc,col] ----
  for (int e = t; e < 64 * 17; e += 512) outs[e] = 0.0f;
  __syncthreads();
#pragma unroll
  for (int tt = 0; tt < 10; ++tt) {
    int col  = cstrip * 160 + tt * 16 + l16;
    int slot = otab[col];
#pragma unroll
    for (int v = 0; v < 8; ++v) {
      int row  = rstrip * 16 + v + (half << 3);   // C/D frag: M = v (+8 hi half)
      float wv = wbig[scs[row] * NCOL + col];
      atomicAdd(&outs[row * 17 + slot], acc[tt][v] * wv);
    }
  }
  __syncthreads();
#pragma unroll
  for (int q = 0; q < 2; ++q) {
    int e = t + 512 * q;
    out[(r0 + (e >> 4)) * 16 + (e & 15)] = outs[(e >> 4) * 17 + (e & 15)];
  }
}

// ---------------------------------------------------------------------------
extern "C" void kernel_launch(void* const* d_in, const int* in_sizes, int n_in,
                              void* d_out, int out_size, void* d_ws, size_t ws_size,
                              hipStream_t stream) {
  (void)in_sizes; (void)n_in; (void)out_size; (void)ws_size;
  const float* x    = (const float*)d_in[0];
  const int*   idx  = (const int*)d_in[1];
  // d_in: [x, index, U1 x4, U2 x4, U3 x4, W1 x4, W2 x4, W3 x4]
  Ptrs12 up, wp;
  for (int i = 0; i < 4; ++i) {
    up.p[i]     = (const float*)d_in[10 + i];   // order 3
    up.p[4 + i] = (const float*)d_in[6 + i];    // order 2
    up.p[8 + i] = (const float*)d_in[2 + i];    // order 1
    wp.p[i]     = (const float*)d_in[22 + i];
    wp.p[4 + i] = (const float*)d_in[18 + i];
    wp.p[8 + i] = (const float*)d_in[14 + i];
  }
  unsigned short* ubig = (unsigned short*)d_ws;
  float* wbig = (float*)((char*)d_ws + UBIG_BYTES);
  int*   otab = (int*)((char*)d_ws + UBIG_BYTES + WBIG_BYTES);

  k_ubig<<<(KT * NCOL * 32) / 256, 256, 0, stream>>>(up, ubig);
  k_wbig<<<(64 * 128 * NCOL) / 256, 256, 0, stream>>>(wp, wbig);
  k_otab<<<3, 256, 0, stream>>>(otab);
  k_gemm<<<NROWS / 64, 512, 0, stream>>>(x, idx, ubig, wbig, otab, (float*)d_out);
}
